// PRoPEAttention_12008728560318
// MI455X (gfx1250) — matrix operand order repaired and checked by reading, not compiled, not run on hardware
//
#include <hip/hip_runtime.h>
#include <hip/hip_bf16.h>

typedef _Float16 f16;
typedef __attribute__((ext_vector_type(16))) _Float16 v16h;
typedef __attribute__((ext_vector_type(8)))  _Float16 v8h;
typedef __attribute__((ext_vector_type(4)))  _Float16 v4h;
typedef __attribute__((ext_vector_type(8)))  float    v8f;
typedef __attribute__((ext_vector_type(4)))  float    v4f;

#define DIM      1024
#define HEADS    16
#define HEAD_DIM 64
#define CAMS     8
#define PER_CAM  256
#define L_TOK    2048
#define B_SZ     2
#define GROUPS   16

__device__ __forceinline__ v16h cat8(v8h a, v8h b) {
  return __builtin_shufflevector(a, b, 0,1,2,3,4,5,6,7,8,9,10,11,12,13,14,15);
}
__device__ __forceinline__ v8f wmma16(v16h a, v16h b, v8f c) {
  return __builtin_amdgcn_wmma_f32_16x16x32_f16(false, a, false, b, (short)0, c, false, false);
}
// A-operand gather from an LDS row: halves j0..7 -> K=hi*8+j, j8..15 -> K=16+hi*8+j
__device__ __forceinline__ v16h ldA(const f16* row, int hi) {
  v8h lo = *(const v8h*)(row + hi * 8);
  v8h hh = *(const v8h*)(row + 16 + hi * 8);
  return cat8(lo, hh);
}

// ---------------------------------------------------------------- cast f32->f16
__global__ void cast_f16_kernel(const float* __restrict__ in, f16* __restrict__ out, int n) {
  int i = blockIdx.x * blockDim.x + threadIdx.x;
  if (i < n) out[i] = (f16)in[i];
}

// ------------------------------------------------- per-camera P, P^-1, P^-T
__global__ void build_cams_kernel(const float* __restrict__ viewmats,
                                  const float* __restrict__ Ks,
                                  float* __restrict__ P, float* __restrict__ Pinv,
                                  float* __restrict__ PinvT) {
  int i = threadIdx.x;
  if (i >= B_SZ * CAMS) return;
  const float* V  = viewmats + i * 16;
  const float* Kc = Ks + i * 9;
  float f = Kc[0], cx = Kc[2], cy = Kc[5];
  float Pm[16];
  #pragma unroll
  for (int c = 0; c < 4; ++c) {
    Pm[0*4+c] = f * V[0*4+c] + cx * V[2*4+c];
    Pm[1*4+c] = f * V[1*4+c] + cy * V[2*4+c];
    Pm[2*4+c] = V[2*4+c];
    Pm[3*4+c] = V[3*4+c];
  }
  // viewmat^-1 = [R^T | -R^T t]
  float Vi[16];
  Vi[0]=V[0]; Vi[1]=V[4]; Vi[2]=V[8];
  Vi[4]=V[1]; Vi[5]=V[5]; Vi[6]=V[9];
  Vi[8]=V[2]; Vi[9]=V[6]; Vi[10]=V[10];
  float t0=V[3], t1=V[7], t2=V[11];
  Vi[3]  = -(Vi[0]*t0 + Vi[1]*t1 + Vi[2]*t2);
  Vi[7]  = -(Vi[4]*t0 + Vi[5]*t1 + Vi[6]*t2);
  Vi[11] = -(Vi[8]*t0 + Vi[9]*t1 + Vi[10]*t2);
  Vi[12]=0.f; Vi[13]=0.f; Vi[14]=0.f; Vi[15]=1.f;
  float invf = 1.0f / f;
  float Ki[16] = { invf,0.f,-cx*invf,0.f,  0.f,invf,-cy*invf,0.f,
                   0.f,0.f,1.f,0.f,        0.f,0.f,0.f,1.f };
  float Pi[16];
  #pragma unroll
  for (int r = 0; r < 4; ++r)
    #pragma unroll
    for (int c = 0; c < 4; ++c) {
      float s = 0.f;
      #pragma unroll
      for (int k = 0; k < 4; ++k) s += Vi[r*4+k] * Ki[k*4+c];
      Pi[r*4+c] = s;
    }
  #pragma unroll
  for (int j = 0; j < 16; ++j) {
    P[i*16+j]     = Pm[j];
    Pinv[i*16+j]  = Pi[j];
    PinvT[i*16+j] = Pi[(j & 3) * 4 + (j >> 2)];
  }
}

// ------------------------------------------------- WMMA GEMM: Y(f32) = X(f16) @ W(f16)^T
// Block tile 64x256, k-step 32, double-buffered LDS (one barrier per k-step).
// 8 waves as 2(M) x 4(N); each wave computes a 32x64 tile = 8 WMMAs per k-step.
#define GBM 64
#define GBN 256
#define GBK 32
#define XLD 40   // LDS row stride in halves (80B: 16B aligned, conflict-friendly)

__global__ __launch_bounds__(256) void gemm_f16_kernel(
    const f16* __restrict__ X, const f16* __restrict__ W,
    float* __restrict__ Y, int M, int N, int K) {
  __shared__ f16 Xs[2][GBM][XLD];   // 10 KB
  __shared__ f16 Ws[2][GBN][XLD];   // 40 KB

  int tid  = threadIdx.x;
  int lane = tid & 31;
  int wid  = tid >> 5;
  int wm   = wid >> 2;          // 0..1 (M)
  int wn   = wid & 3;           // 0..3 (N)
  int n    = lane & 15;
  int hi   = lane >> 4;
  int row0 = blockIdx.y * GBM;
  int col0 = blockIdx.x * GBN;

  // loader mapping: v8h per thread; X: 64x32 halves, W: 4 x (64x32) halves
  int lr = tid >> 2;            // 0..63
  int lc = (tid & 3) * 8;       // 0,8,16,24
  const f16* Xp = X + (size_t)(row0 + lr) * K + lc;
  const f16* Wp = W + (size_t)(col0 + lr) * K + lc;

  // preload tile 0 into buffer 0
  {
    *(v8h*)&Xs[0][lr][lc] = *(const v8h*)Xp;
    #pragma unroll
    for (int i = 0; i < 4; ++i)
      *(v8h*)&Ws[0][lr + 64*i][lc] = *(const v8h*)(Wp + (size_t)(64*i) * K);
  }
  __syncthreads();

  v8f acc[2][4] = {};
  int nk = K / GBK;
  for (int kk = 0; kk < nk; ++kk) {
    int buf = kk & 1;
    bool more = (kk + 1) < nk;          // uniform across block
    v8h xv; v8h wv[4];
    if (more) {                          // issue next-tile global loads early
      const f16* xp = Xp + (kk + 1) * GBK;
      const f16* wp = Wp + (kk + 1) * GBK;
      xv = *(const v8h*)xp;
      #pragma unroll
      for (int i = 0; i < 4; ++i) wv[i] = *(const v8h*)(wp + (size_t)(64*i) * K);
    }

    // compute from current buffer
    v16h a0 = ldA(&Xs[buf][wm*32 + n][0], hi);
    v16h a1 = ldA(&Xs[buf][wm*32 + 16 + n][0], hi);
    #pragma unroll
    for (int j = 0; j < 4; ++j) {
      const f16* bp = &Ws[buf][wn*64 + j*16 + n][hi*8];
      v16h b = cat8(*(const v8h*)bp, *(const v8h*)(bp + 16));
      acc[0][j] = wmma16(a0, b, acc[0][j]);
      acc[1][j] = wmma16(a1, b, acc[1][j]);
    }

    if (more) {                          // stage next tile into the other buffer
      *(v8h*)&Xs[buf ^ 1][lr][lc] = xv;
      #pragma unroll
      for (int i = 0; i < 4; ++i) *(v8h*)&Ws[buf ^ 1][lr + 64*i][lc] = wv[i];
    }
    __syncthreads();
  }

  #pragma unroll
  for (int at = 0; at < 2; ++at)
    #pragma unroll
    for (int j = 0; j < 4; ++j)
      #pragma unroll
      for (int r = 0; r < 8; ++r)
        Y[(size_t)(row0 + wm*32 + at*16 + r + 8*hi) * N +
          (col0 + wn*64 + j*16 + n)] = acc[at][j][r];
}

// ------------------------------------------------- PRoPE transform of q,k,v
// q' = P^-T (D q) * 1/sqrt(64),  k' = P (D k),  v' = P (D v); output f16 [B][H][L][64]
__global__ void transform_qkv_kernel(const float* __restrict__ qkv,
                                     const float* __restrict__ P,
                                     const float* __restrict__ PinvT,
                                     f16* __restrict__ q, f16* __restrict__ k,
                                     f16* __restrict__ v) {
  int idx = blockIdx.x * blockDim.x + threadIdx.x;  // b:1 | l:11 | h:4 | g:4
  if (idx >= B_SZ * L_TOK * HEADS * GROUPS) return;
  int g = idx & 15;
  int h = (idx >> 4) & 15;
  int l = (idx >> 8) & (L_TOK - 1);
  int b = idx >> 19;
  int cam = l >> 8;
  int p   = l & 255;

  float u  = ((p & 15) + 0.5f) * 16.0f;
  float vv = ((p >> 4) + 0.5f) * 16.0f;
  float fr = __powf(10000.0f, -(float)g * (1.0f / 16.0f));
  float sa, ca, sb, cb;
  __sincosf(u * fr, &sa, &ca);
  __sincosf(vv * fr, &sb, &cb);

  const float* Pm  = P     + (b * CAMS + cam) * 16;
  const float* PiT = PinvT + (b * CAMS + cam) * 16;
  size_t rowbase = ((size_t)b * L_TOK + l) * (3 * DIM) + h * HEAD_DIM + g * 4;
  size_t obase   = (((size_t)(b * HEADS + h)) * L_TOK + l) * HEAD_DIM + g * 4;

  #pragma unroll
  for (int s = 0; s < 3; ++s) {
    v4f t = *(const v4f*)(qkv + rowbase + (size_t)s * DIM);
    float r0 = ca * t[0] - sa * t[1];
    float r1 = sa * t[0] + ca * t[1];
    float r2 = cb * t[2] - sb * t[3];
    float r3 = sb * t[2] + cb * t[3];
    const float* M = (s == 0) ? PiT : Pm;
    float sc = (s == 0) ? 0.125f : 1.0f;  // fold softmax 1/sqrt(64) into q
    v4h o;
    #pragma unroll
    for (int i = 0; i < 4; ++i)
      o[i] = (f16)((M[i*4+0]*r0 + M[i*4+1]*r1 + M[i*4+2]*r2 + M[i*4+3]*r3) * sc);
    f16* dst = (s == 0) ? q : ((s == 1) ? k : v);
    *(v4h*)(dst + obase) = o;
  }
}

// ------------------------------------------------- flash attention (WMMA, wave32)
// Per wave: 16 queries. Chunk = 32 keys staged in LDS (K row-major, V transposed).
// S^T = K_tile @ Q^T ; online softmax on S^T ; O^T += V^T @ P^T.
#define ACK 32
__global__ __launch_bounds__(256) void attn_kernel(
    const f16* __restrict__ Qg, const f16* __restrict__ Kg,
    const f16* __restrict__ Vg, float* __restrict__ Og) {
  __shared__ f16 Ksh[ACK][72];
  __shared__ f16 Vt[HEAD_DIM][40];

  int bh = blockIdx.z * HEADS + blockIdx.y;
  const f16* Qb = Qg + (size_t)bh * L_TOK * HEAD_DIM;
  const f16* Kb = Kg + (size_t)bh * L_TOK * HEAD_DIM;
  const f16* Vb = Vg + (size_t)bh * L_TOK * HEAD_DIM;
  float*     Ob = Og + (size_t)bh * L_TOK * HEAD_DIM;

  int tid  = threadIdx.x;
  int lane = tid & 31;
  int wid  = tid >> 5;
  int n    = lane & 15;
  int hi   = lane >> 4;
  int q0   = blockIdx.x * 128 + wid * 16;

  // Q as B-operand: b[j] = Q[n][dbase + hi*8 + j] (j<8), Q[n][dbase + 16 + hi*8 + j-8] (j>=8)
  const f16* qrow = Qb + (size_t)(q0 + n) * HEAD_DIM;
  v16h bq0 = cat8(*(const v8h*)(qrow + hi * 8), *(const v8h*)(qrow + 16 + hi * 8));
  v16h bq1 = cat8(*(const v8h*)(qrow + 32 + hi * 8), *(const v8h*)(qrow + 48 + hi * 8));

  v8f ot[4] = {};
  float mprev = -1e30f, lsum = 0.f;

  for (int k0 = 0; k0 < L_TOK; k0 += ACK) {
    __syncthreads();
    { // stage K (row-major) and V (transposed) chunks
      int r = tid >> 3, d0 = (tid & 7) * 8;
      *(v8h*)&Ksh[r][d0] = *(const v8h*)(Kb + (size_t)(k0 + r) * HEAD_DIM + d0);
      v8h vv = *(const v8h*)(Vb + (size_t)(k0 + r) * HEAD_DIM + d0);
      #pragma unroll
      for (int e = 0; e < 8; ++e) Vt[d0 + e][r] = vv[e];
      if (k0 + ACK < L_TOK) {   // prefetch next chunk (global_prefetch_b8)
        __builtin_prefetch(Kb + (size_t)(k0 + ACK + r) * HEAD_DIM + d0, 0, 0);
        __builtin_prefetch(Vb + (size_t)(k0 + ACK + r) * HEAD_DIM + d0, 0, 0);
      }
    }
    __syncthreads();

    // S^T tiles (keys x queries), f32 accum
    v8f st[2];
    #pragma unroll
    for (int t = 0; t < 2; ++t) {
      const f16* krow = &Ksh[t * 16 + n][0];
      v8f c = {};
      c = wmma16(ldA(krow, hi),      bq0, c);
      c = wmma16(ldA(krow + 32, hi), bq1, c);
      st[t] = c;
    }

    // online softmax over this 32-key chunk (per query column n)
    float cmax = -1e30f;
    #pragma unroll
    for (int r = 0; r < 8; ++r) cmax = fmaxf(cmax, fmaxf(st[0][r], st[1][r]));
    cmax = fmaxf(cmax, __shfl_xor(cmax, 16, 32));
    float mnew  = fmaxf(mprev, cmax);
    float alpha = __expf(mprev - mnew);
    float p0[8], p1[8], csum = 0.f;
    #pragma unroll
    for (int r = 0; r < 8; ++r) {
      p0[r] = __expf(st[0][r] - mnew);
      p1[r] = __expf(st[1][r] - mnew);
      csum += p0[r] + p1[r];
    }
    csum += __shfl_xor(csum, 16, 32);
    lsum  = lsum * alpha + csum;
    mprev = mnew;
    #pragma unroll
    for (int t = 0; t < 4; ++t) ot[t] *= alpha;

    // repack P^T (C layout) into B-operand layout with one half-wave exchange
    v16h bp;
    #pragma unroll
    for (int r = 0; r < 8; ++r) {
      float send = hi ? p0[r] : p1[r];
      float rec  = __shfl_xor(send, 16, 32);
      float lo   = hi ? rec   : p0[r];
      float hh   = hi ? p1[r] : rec;
      bp[r]     = (f16)lo;
      bp[8 + r] = (f16)hh;
    }

    // O^T += V^T @ P^T  (4 d-tiles of 16)
    #pragma unroll
    for (int dt = 0; dt < 4; ++dt)
      ot[dt] = wmma16(cat8(*(const v8h*)&Vt[dt * 16 + n][hi * 16], *(const v8h*)&Vt[dt * 16 + n][hi * 16 + 8]), bp, ot[dt]);
  }

  float inv_l = 1.0f / lsum;
  #pragma unroll
  for (int dt = 0; dt < 4; ++dt)
    #pragma unroll
    for (int r = 0; r < 8; ++r)
      Ob[(size_t)(q0 + n) * HEAD_DIM + dt * 16 + 8 * hi + r] = ot[dt][r] * inv_l;
}

// ------------------------------------------------- output transform: o' = D^T (P^-1 o)
__global__ void transform_o_kernel(const float* __restrict__ Obuf,
                                   const float* __restrict__ Pinv,
                                   f16* __restrict__ O2) {
  int idx = blockIdx.x * blockDim.x + threadIdx.x;  // b:1 | h:4 | l:11 | g:4
  if (idx >= B_SZ * HEADS * L_TOK * GROUPS) return;
  int g = idx & 15;
  int l = (idx >> 4) & (L_TOK - 1);
  int h = (idx >> 15) & 15;
  int b = idx >> 19;
  int cam = l >> 8;
  int p   = l & 255;

  float u  = ((p & 15) + 0.5f) * 16.0f;
  float vv = ((p >> 4) + 0.5f) * 16.0f;
  float fr = __powf(10000.0f, -(float)g * (1.0f / 16.0f));
  float sa, ca, sb, cb;
  __sincosf(u * fr, &sa, &ca);
  __sincosf(vv * fr, &sb, &cb);

  const float* Pi = Pinv + (b * CAMS + cam) * 16;
  v4f t = *(const v4f*)(Obuf + (size_t)idx * 4);
  float w[4];
  #pragma unroll
  for (int i = 0; i < 4; ++i)
    w[i] = Pi[i*4+0]*t[0] + Pi[i*4+1]*t[1] + Pi[i*4+2]*t[2] + Pi[i*4+3]*t[3];
  v4h o;
  o[0] = (f16)( ca * w[0] + sa * w[1]);
  o[1] = (f16)(-sa * w[0] + ca * w[1]);
  o[2] = (f16)( cb * w[2] + sb * w[3]);
  o[3] = (f16)(-sb * w[2] + cb * w[3]);
  size_t dst = ((size_t)b * L_TOK + l) * DIM + h * HEAD_DIM + g * 4;
  *(v4h*)(O2 + dst) = o;
}

// ----------------------------------------------------------------------------
extern "C" void kernel_launch(void* const* d_in, const int* in_sizes, int n_in,
                              void* d_out, int out_size, void* d_ws, size_t ws_size,
                              hipStream_t stream) {
  const float* x        = (const float*)d_in[0];
  const float* viewmats = (const float*)d_in[1];
  const float* Ks       = (const float*)d_in[2];
  const float* w_qkv    = (const float*)d_in[3];
  const float* w_proj   = (const float*)d_in[4];
  float* out = (float*)d_out;
  char*  ws  = (char*)d_ws;

  const int ML = B_SZ * L_TOK;           // 4096
  // workspace layout (all 256B aligned)
  size_t o_xh    = 0;                    // 4096*1024 f16
  size_t o_wqkv  = o_xh    + (size_t)ML * DIM * 2;
  size_t o_wproj = o_wqkv  + (size_t)3 * DIM * DIM * 2;
  size_t o_qkv   = o_wproj + (size_t)DIM * DIM * 2;
  size_t o_qh    = o_qkv   + (size_t)ML * 3 * DIM * 4;
  size_t o_kh    = o_qh    + (size_t)ML * DIM * 2;
  size_t o_vh    = o_kh    + (size_t)ML * DIM * 2;
  size_t o_ob    = o_vh    + (size_t)ML * DIM * 2;
  size_t o_o2    = o_ob    + (size_t)ML * DIM * 4;
  size_t o_P     = o_o2    + (size_t)ML * DIM * 2;
  size_t o_Pinv  = o_P     + 1024;
  size_t o_PinvT = o_Pinv  + 1024;

  f16*   xh    = (f16*)(ws + o_xh);
  f16*   wqkvh = (f16*)(ws + o_wqkv);
  f16*   wprojh= (f16*)(ws + o_wproj);
  float* qkv   = (float*)(ws + o_qkv);
  f16*   qh    = (f16*)(ws + o_qh);
  f16*   kh    = (f16*)(ws + o_kh);
  f16*   vh    = (f16*)(ws + o_vh);
  float* obuf  = (float*)(ws + o_ob);
  f16*   o2h   = (f16*)(ws + o_o2);
  float* Pd    = (float*)(ws + o_P);
  float* Pinvd = (float*)(ws + o_Pinv);
  float* PinvTd= (float*)(ws + o_PinvT);

  int nx = ML * DIM, nwq = 3 * DIM * DIM, nwp = DIM * DIM;
  cast_f16_kernel<<<(nx  + 255) / 256, 256, 0, stream>>>(x, xh, nx);
  cast_f16_kernel<<<(nwq + 255) / 256, 256, 0, stream>>>(w_qkv, wqkvh, nwq);
  cast_f16_kernel<<<(nwp + 255) / 256, 256, 0, stream>>>(w_proj, wprojh, nwp);
  build_cams_kernel<<<1, 32, 0, stream>>>(viewmats, Ks, Pd, Pinvd, PinvTd);

  // qkv = x @ w_qkv^T : 4096 x 3072, K=1024
  gemm_f16_kernel<<<dim3(3 * DIM / GBN, ML / GBM), 256, 0, stream>>>(
      xh, wqkvh, qkv, ML, 3 * DIM, DIM);

  transform_qkv_kernel<<<(B_SZ * L_TOK * HEADS * GROUPS) / 256, 256, 0, stream>>>(
      qkv, Pd, PinvTd, qh, kh, vh);

  attn_kernel<<<dim3(L_TOK / 128, HEADS, B_SZ), 256, 0, stream>>>(qh, kh, vh, obuf);

  transform_o_kernel<<<(B_SZ * HEADS * L_TOK * GROUPS) / 256, 256, 0, stream>>>(
      obuf, Pinvd, o2h);

  // out = o2 @ w_proj^T : 4096 x 1024, K=1024
  gemm_f16_kernel<<<dim3(DIM / GBN, ML / GBM), 256, 0, stream>>>(
      o2h, wprojh, out, ML, DIM, DIM);
}
